// SLoRALinear_55001351193152
// MI455X (gfx1250) — matrix operand order repaired and checked by reading, not compiled, not run on hardware
//
#include <hip/hip_runtime.h>

// SLoRA fused linear for MI455X (gfx1250, wave32, WMMA).
//   out[b,t,o] = x[b,t,:] . W[o,:]  +  s_b * (x[b,t,:] . A[a]^T) . Bm[a][o,:]
//
// Precision strategy: the 32768x2048x2048 GEMM is matrix-unit bound
// (AI ~500 FLOP/B; HBM time ~35us at 23.3 TB/s), so we use the chip's fast
// path, V_WMMA_F32_16X16X32_BF16, with bf16x3 split emulation for fp32-grade
// accuracy:  x = hi + lo (hi=bf16(x), lo=bf16(x-hi));
//            acc += hi*Hi + hi*Lo + lo*Hi   (fp32 accumulate, drop lo*lo).
// 3 bf16 WMMAs replace 8 f32 WMMAs per K=32 -> ~2.7x effective throughput.
// The split is done once at LDS staging (amortized over 128 reuses).
//
//   Pass 1: XA = s_b * x_b @ A[a]^T  -> (B*T, 16) in workspace (2 MB);
//           memory-bound, kept on the exact V_WMMA_F32_16X16X4_F32 path.
//   Pass 2: 128x128 macro-tile GEMM out = x @ W^T with a fused zero-padded
//           K=32 bf16 LoRA chunk (XA @ Bm[a]^T); adapter uniform per block
//           since 128 | T.

typedef float  v2f  __attribute__((ext_vector_type(2)));
typedef float  v8f  __attribute__((ext_vector_type(8)));
typedef __bf16 bf16_t;
typedef bf16_t v4bf  __attribute__((ext_vector_type(4)));
typedef bf16_t v8bf  __attribute__((ext_vector_type(8)));
typedef bf16_t v16bf __attribute__((ext_vector_type(16)));

#define D_IN   2048
#define D_OUT  2048
#define RANK   16
#define BM     128
#define BN     128
#define BK     32
#define LDT    36   // pass-1 f32 LDS pitch (floats): 144B rows
#define LDB    40   // pass-2 bf16 LDS pitch (elems): 80B rows, 16B aligned;
                    // 16-lane b128 frag loads start at distinct 4-bank slots

// ---------------------------------------------------------------- pass 1 ---
__global__ __launch_bounds__(256) void slora_xa_kernel(
    const float* __restrict__ X,      // (B,T,D_IN)
    const float* __restrict__ A,      // (NA, RANK, D_IN)
    const float* __restrict__ scal,   // (NA,)
    const int*   __restrict__ aidx,   // (B,)
    float*       __restrict__ XA)     // (B*T, RANK)
{
    __shared__ float As[BM * LDT];
    __shared__ float Bs[RANK * LDT];

    const int tid    = threadIdx.x;
    const int mb     = blockIdx.x;          // 0..255
    const int batch  = mb >> 4;
    const int blockM = mb * BM;
    const int wave   = tid >> 5;
    const int lane   = tid & 31;
    const int lrow   = lane & 15;
    const int lhi    = lane >> 4;

    const int   a = aidx[batch];
    const float s = scal[a];
    const float* Ap = A + (size_t)a * RANK * D_IN;

    v8f acc = {};

    for (int k0 = 0; k0 < D_IN; k0 += BK) {
#pragma unroll
        for (int i = 0; i < 4; ++i) {
            int f = i * 256 + tid;
            int row = f >> 3, c4 = (f & 7) << 2;
            float4 v = *(const float4*)(X + (size_t)(blockM + row) * D_IN + k0 + c4);
            *(float4*)(&As[row * LDT + c4]) = v;
        }
        if (tid < 128) {
            int row = tid >> 3, c4 = (tid & 7) << 2;
            float4 v = *(const float4*)(Ap + (size_t)row * D_IN + k0 + c4);
            *(float4*)(&Bs[row * LDT + c4]) = v;
        }
        __syncthreads();
#pragma unroll
        for (int kk = 0; kk < BK; kk += 4) {
            v2f af = *(const v2f*)(&As[(wave * 16 + lrow) * LDT + kk + lhi * 2]);
            v2f bf = *(const v2f*)(&Bs[lrow * LDT + kk + lhi * 2]);
            acc = __builtin_amdgcn_wmma_f32_16x16x4_f32(
                false, af, false, bf, (short)0, acc, false, false);
        }
        __syncthreads();
    }
#pragma unroll
    for (int v = 0; v < 8; ++v) {
        int row = blockM + wave * 16 + v + lhi * 8;
        XA[(size_t)row * RANK + lrow] = acc[v] * s;   // pre-scale by s_b
    }
}

// ------------------------------------------------------------- helpers ----
__device__ __forceinline__ void split_store4(bf16_t* __restrict__ hiArr,
                                             bf16_t* __restrict__ loArr,
                                             int idx, float4 v) {
    bf16_t h0 = (bf16_t)v.x, h1 = (bf16_t)v.y, h2 = (bf16_t)v.z, h3 = (bf16_t)v.w;
    bf16_t l0 = (bf16_t)(v.x - (float)h0);
    bf16_t l1 = (bf16_t)(v.y - (float)h1);
    bf16_t l2 = (bf16_t)(v.z - (float)h2);
    bf16_t l3 = (bf16_t)(v.w - (float)h3);
    *(v4bf*)(&hiArr[idx]) = (v4bf){h0, h1, h2, h3};
    *(v4bf*)(&loArr[idx]) = (v4bf){l0, l1, l2, l3};
}

__device__ __forceinline__ v16bf load_frag(const bf16_t* __restrict__ p,
                                           int o1, int o2) {
    v8bf a = *(const v8bf*)(p + o1);
    v8bf b = *(const v8bf*)(p + o2);
    return __builtin_shufflevector(a, b, 0, 1, 2, 3, 4, 5, 6, 7,
                                         8, 9, 10, 11, 12, 13, 14, 15);
}

// ---------------------------------------------------------------- pass 2 ---
// Grid (16 N-tiles, 256 M-tiles). 8 waves (4 in M x 2 in N); each wave owns a
// 32x64 tile = 2x4 f32 accumulators. Per K=32 chunk: 24 ds_load_b128 frag
// loads + 24 v_wmma_f32_16x16x32_bf16 (bf16x3 emulation).
__global__ __launch_bounds__(256) void slora_main_kernel(
    const float* __restrict__ X,      // (B*T, D_IN)
    const float* __restrict__ W,      // (D_OUT, D_IN)
    const float* __restrict__ Bmat,   // (NA, D_OUT, RANK)
    const float* __restrict__ XA,     // (B*T, RANK), pre-scaled
    const int*   __restrict__ aidx,   // (B,)
    float*       __restrict__ Out)    // (B*T, D_OUT)
{
    __shared__ bf16_t AsH[BM * LDB], AsL[BM * LDB];
    __shared__ bf16_t BsH[BN * LDB], BsL[BN * LDB];

    const int tid    = threadIdx.x;
    const int nb     = blockIdx.x;          // 0..15
    const int mb     = blockIdx.y;          // 0..255
    const int batch  = mb >> 4;
    const int blockM = mb * BM;
    const int blockN = nb * BN;
    const int wave   = tid >> 5;
    const int lane   = tid & 31;
    const int lrow   = lane & 15;
    const int lhi    = lane >> 4;
    const int waveM  = wave & 3;            // 4 waves x 32 rows
    const int waveN  = wave >> 2;           // 2 waves x 64 cols

    // fragment offsets within a row (elements):
    //   A (16x32 16-bit layout): lanes<16 K0-7 || K16-23 ; lanes>=16 +8
    //   B (32x16):               lanes<16 K0-7 || K16-23 ; lanes>=16 +8
    const int aO1 = lhi * 8,  aO2 = 16 + lhi * 8;
    const int bO1 = lhi * 8,  bO2 = 16 + lhi * 8;

    v8f acc[2][4] = {};

    for (int k0 = 0; k0 < D_IN; k0 += BK) {
#pragma unroll
        for (int i = 0; i < 4; ++i) {       // x tile -> split hi/lo
            int f = i * 256 + tid;
            int row = f >> 3, c4 = (f & 7) << 2;
            float4 v = *(const float4*)(X + (size_t)(blockM + row) * D_IN + k0 + c4);
            split_store4(AsH, AsL, row * LDB + c4, v);
        }
#pragma unroll
        for (int i = 0; i < 4; ++i) {       // W tile -> split hi/lo
            int f = i * 256 + tid;
            int row = f >> 3, c4 = (f & 7) << 2;
            float4 v = *(const float4*)(W + (size_t)(blockN + row) * D_IN + k0 + c4);
            split_store4(BsH, BsL, row * LDB + c4, v);
        }
        __syncthreads();

        v16bf bh[4], bl[4];
#pragma unroll
        for (int ni = 0; ni < 4; ++ni) {
            int r = (waveN * 64 + ni * 16 + lrow) * LDB;
            bh[ni] = load_frag(&BsH[r], bO1, bO2);
            bl[ni] = load_frag(&BsL[r], bO1, bO2);
        }
#pragma unroll
        for (int mi = 0; mi < 2; ++mi) {
            int r = (waveM * 32 + mi * 16 + lrow) * LDB;
            v16bf ah = load_frag(&AsH[r], aO1, aO2);
            v16bf al = load_frag(&AsL[r], aO1, aO2);
#pragma unroll
            for (int ni = 0; ni < 4; ++ni) {
                acc[mi][ni] = __builtin_amdgcn_wmma_f32_16x16x32_bf16(
                    false, ah, false, bh[ni], (short)0, acc[mi][ni], false, false);
                acc[mi][ni] = __builtin_amdgcn_wmma_f32_16x16x32_bf16(
                    false, ah, false, bl[ni], (short)0, acc[mi][ni], false, false);
                acc[mi][ni] = __builtin_amdgcn_wmma_f32_16x16x32_bf16(
                    false, al, false, bh[ni], (short)0, acc[mi][ni], false, false);
            }
        }
        __syncthreads();
    }

    // ---- LoRA epilogue: one zero-padded K=32 chunk of [s*xa] @ Bm[a]^T ----
    {
        const int a = aidx[batch];
        const float* Bp = Bmat + (size_t)a * D_OUT * RANK;
#pragma unroll
        for (int i = 0; i < 2; ++i) {       // XA tile: 128x16 into cols 0..15
            int f = i * 256 + tid;
            int row = f >> 2, c4 = (f & 3) << 2;
            float4 v = *(const float4*)(XA + (size_t)(blockM + row) * RANK + c4);
            split_store4(AsH, AsL, row * LDB + c4, v);
        }
#pragma unroll
        for (int i = 0; i < 2; ++i) {       // Bm tile: 128x16 into cols 0..15
            int f = i * 256 + tid;
            int row = f >> 2, c4 = (f & 3) << 2;
            float4 v = *(const float4*)(Bp + (size_t)(blockN + row) * RANK + c4);
            split_store4(BsH, BsL, row * LDB + c4, v);
        }
        v4bf z = {};
#pragma unroll
        for (int i = 0; i < 2; ++i) {       // zero cols 16..31
            int f = i * 256 + tid;
            int row = f >> 2, c4 = ((f & 3) << 2) + 16;
            *(v4bf*)(&AsH[row * LDB + c4]) = z;
            *(v4bf*)(&AsL[row * LDB + c4]) = z;
            *(v4bf*)(&BsH[row * LDB + c4]) = z;
            *(v4bf*)(&BsL[row * LDB + c4]) = z;
        }
        __syncthreads();

        v16bf bh[4], bl[4];
#pragma unroll
        for (int ni = 0; ni < 4; ++ni) {
            int r = (waveN * 64 + ni * 16 + lrow) * LDB;
            bh[ni] = load_frag(&BsH[r], bO1, bO2);
            bl[ni] = load_frag(&BsL[r], bO1, bO2);
        }
#pragma unroll
        for (int mi = 0; mi < 2; ++mi) {
            int r = (waveM * 32 + mi * 16 + lrow) * LDB;
            v16bf ah = load_frag(&AsH[r], aO1, aO2);
            v16bf al = load_frag(&AsL[r], aO1, aO2);
#pragma unroll
            for (int ni = 0; ni < 4; ++ni) {
                acc[mi][ni] = __builtin_amdgcn_wmma_f32_16x16x32_bf16(
                    false, ah, false, bh[ni], (short)0, acc[mi][ni], false, false);
                acc[mi][ni] = __builtin_amdgcn_wmma_f32_16x16x32_bf16(
                    false, ah, false, bl[ni], (short)0, acc[mi][ni], false, false);
                acc[mi][ni] = __builtin_amdgcn_wmma_f32_16x16x32_bf16(
                    false, al, false, bh[ni], (short)0, acc[mi][ni], false, false);
            }
        }
    }

    // ---- store per documented C/D layout: VGPR v -> rows v / v+8 ----
#pragma unroll
    for (int mi = 0; mi < 2; ++mi)
#pragma unroll
        for (int ni = 0; ni < 4; ++ni)
#pragma unroll
            for (int v = 0; v < 8; ++v) {
                int row = blockM + waveM * 32 + mi * 16 + v + lhi * 8;
                int col = blockN + waveN * 64 + ni * 16 + lrow;
                Out[(size_t)row * D_OUT + col] = acc[mi][ni][v];
            }
}

extern "C" void kernel_launch(void* const* d_in, const int* in_sizes, int n_in,
                              void* d_out, int out_size, void* d_ws, size_t ws_size,
                              hipStream_t stream) {
    const float* x    = (const float*)d_in[0];   // (16, 2048, 2048) f32
    const float* W    = (const float*)d_in[1];   // (2048, 2048) f32
    const float* A    = (const float*)d_in[2];   // (8, 16, 2048) f32
    const float* Bm   = (const float*)d_in[3];   // (8, 2048, 16) f32
    const float* scal = (const float*)d_in[4];   // (8,) f32
    const int*   aidx = (const int*)d_in[5];     // (16,) i32
    float* out = (float*)d_out;                  // (16, 2048, 2048) f32
    float* XA  = (float*)d_ws;                   // (32768, 16) f32 = 2 MB

    slora_xa_kernel<<<dim3(256), dim3(256), 0, stream>>>(x, A, scal, aidx, XA);
    slora_main_kernel<<<dim3(16, 256), dim3(256), 0, stream>>>(x, W, Bm, XA,
                                                               aidx, out);
}
